// DependencyParser_6330781795078
// MI455X (gfx1250) — matrix operand order repaired and checked by reading, not compiled, not run on hardware
//
#include <hip/hip_runtime.h>
#include <hip/hip_bf16.h>

typedef _Float16 v16h __attribute__((ext_vector_type(16)));
typedef float    v8f  __attribute__((ext_vector_type(8)));

#define L_SEQ 512
#define H_DIM 512
#define G_DIM 2048   // 4*H
#define DIN0  320
#define DIN1  1024
#define M_DIM 512

__device__ __forceinline__ float fast_sigmoid(float x) {
    return 1.0f / (1.0f + __expf(-x));
}
__device__ __forceinline__ float fast_tanh(float x) {
    return 2.0f / (1.0f + __expf(-2.0f * x)) - 1.0f;
}

// ---------------------------------------------------------------------------
// WMMA 16-bit A-operand packed layout.
// Forward map (lane l, half h, tile kt):
//   kloc = h + 16*(l>=16)
// packA_idx inverts it: (row, k) -> flat index into packed buffer, where the
// buffer is [mtiles][KT][32 lanes][16 halves] so a lane reads 32 contiguous B.
// ---------------------------------------------------------------------------
__device__ __forceinline__ int packA_idx(int row, int k, int KT) {
    int mt    = row >> 4, mr = row & 15;
    int kt    = k >> 5,  kloc = k & 31;
    int lhalf = kloc >> 4;
    int h     = kloc & 15;
    int lane  = mr + (lhalf << 4);
    return ((mt * KT + kt) * 32 + lane) * 16 + h;
}

// ---------------------------------------------------------------------------
// Embedding gather -> A-packed f16 activations x[512][320]
// ---------------------------------------------------------------------------
__global__ void embed_kernel(const int* __restrict__ widx, const int* __restrict__ pidx,
                             const float* __restrict__ wemb, const float* __restrict__ pemb,
                             _Float16* __restrict__ x_pk) {
    int row = blockIdx.x;          // 0..511
    int c   = threadIdx.x;         // 0..319
    int wi  = widx[row];
    int pi  = pidx[row];
    float v = (c < 256) ? wemb[wi * 256 + c] : pemb[pi * 64 + (c - 256)];
    x_pk[packA_idx(row, c, DIN0 / 32)] = (_Float16)v;
}

// ---------------------------------------------------------------------------
// Elementwise f32 -> f16 (row-major weight copies for the WMMA B operand)
// ---------------------------------------------------------------------------
__global__ void f32_to_f16_kernel(const float* __restrict__ src,
                                  _Float16* __restrict__ dst, int n) {
    int i = blockIdx.x * 256 + threadIdx.x;
    if (i < n) dst[i] = (_Float16)src[i];
}

// ---------------------------------------------------------------------------
// Pack W_hh [2048,512] f32 into WMMA A-operand f16 layout (used 512x in the
// recurrence; f16 halves L2 traffic, packing makes lane loads contiguous 32B).
// ---------------------------------------------------------------------------
__global__ void pack_whh_kernel(const float* __restrict__ w0, const float* __restrict__ w1,
                                const float* __restrict__ w2, const float* __restrict__ w3,
                                _Float16* __restrict__ out) {
    int mat = blockIdx.y;
    const float* W = (mat == 0) ? w0 : (mat == 1) ? w1 : (mat == 2) ? w2 : w3;
    _Float16* dst = out + (size_t)mat * (G_DIM * H_DIM);
    int d  = blockIdx.x * blockDim.x + threadIdx.x;   // 0..2048*512-1
    int h  = d & 15;
    int l  = (d >> 4) & 31;
    int kt = (d >> 9) & 15;
    int rt = d >> 13;                                  // 0..127
    int v  = h >> 1, hi = h & 1;
    int kloc = h + ((l >> 4) << 4);
    int row = rt * 16 + (l & 15);
    int k   = kt * 32 + kloc;
    dst[d] = (_Float16)W[row * H_DIM + k];
}

// ---------------------------------------------------------------------------
// WMMA GEMM: C[512,N] f32 = Apk(512,K f16, A-packed) @ Wh[N,K]^T (f16 row-major)
//            + bias1 + bias2.
// Block = 256 thr = 8 waves; wave owns one 16-row M tile x four 16-col N tiles.
// k-loop unrolled x2 with ping-pong operand buffers (a0/b0 vs a1/b1): compute
// from one buffer while prefetching the other -> partial loadcnt waits, and no
// register-rotation movs in the loop body.
// grid = (512/16/8, N/64); KT is even (10 or 32).
// ---------------------------------------------------------------------------
__global__ __launch_bounds__(256)
void gemm_wmma_kernel(const _Float16* __restrict__ Apk, const _Float16* __restrict__ Wh,
                      const float* __restrict__ b1p, const float* __restrict__ b2p,
                      float* __restrict__ C, int K, int N) {
    int tid   = threadIdx.x;
    int l     = tid & 31;
    int w     = tid >> 5;
    int lhalf = l >> 4;
    int lmod  = l & 15;
    int mt    = blockIdx.x * 8 + w;        // M tile (0..31)
    int ntb   = blockIdx.y * 4;            // first of 4 N tiles
    int KT    = K >> 5;

    v8f acc[4];
#pragma unroll
    for (int nt = 0; nt < 4; ++nt) {
        int n = (ntb + nt) * 16 + lmod;
        float bias = 0.0f;
        if (b1p) bias += b1p[n];
        if (b2p) bias += b2p[n];
#pragma unroll
        for (int v = 0; v < 8; ++v) acc[nt][v] = bias;
    }

    const _Float16* ap = Apk + (((size_t)mt * KT) << 9) + (l << 4);  // tile = 512 halves
    const _Float16* wp[4];
#pragma unroll
    for (int nt = 0; nt < 4; ++nt)
        wp[nt] = Wh + (size_t)((ntb + nt) * 16 + lmod) * K + (lhalf << 4);

    v16h a0, a1, b0[4], b1[4];
    // prologue: load kt = 0 into buffer 0
    a0 = *(const v16h*)(ap);
#pragma unroll
    for (int nt = 0; nt < 4; ++nt) b0[nt] = *(const v16h*)(wp[nt]);

    int kt = 0;
    for (; kt + 2 < KT; kt += 2) {
        // prefetch kt+1 into buffer 1
        a1 = *(const v16h*)(ap + ((kt + 1) << 9));
#pragma unroll
        for (int nt = 0; nt < 4; ++nt) b1[nt] = *(const v16h*)(wp[nt] + ((kt + 1) << 5));
        // compute kt from buffer 0
#pragma unroll
        for (int nt = 0; nt < 4; ++nt)
            acc[nt] = __builtin_amdgcn_wmma_f32_16x16x32_f16(false, a0, false, b0[nt],
                                                             (short)0, acc[nt], false, false);
        // prefetch kt+2 into buffer 0
        a0 = *(const v16h*)(ap + ((kt + 2) << 9));
#pragma unroll
        for (int nt = 0; nt < 4; ++nt) b0[nt] = *(const v16h*)(wp[nt] + ((kt + 2) << 5));
        // compute kt+1 from buffer 1
#pragma unroll
        for (int nt = 0; nt < 4; ++nt)
            acc[nt] = __builtin_amdgcn_wmma_f32_16x16x32_f16(false, a1, false, b1[nt],
                                                             (short)0, acc[nt], false, false);
    }
    // epilogue: kt == KT-2 (buffer 0 holds it); load KT-1 then compute both
    a1 = *(const v16h*)(ap + ((KT - 1) << 9));
#pragma unroll
    for (int nt = 0; nt < 4; ++nt) b1[nt] = *(const v16h*)(wp[nt] + ((KT - 1) << 5));
#pragma unroll
    for (int nt = 0; nt < 4; ++nt)
        acc[nt] = __builtin_amdgcn_wmma_f32_16x16x32_f16(false, a0, false, b0[nt],
                                                         (short)0, acc[nt], false, false);
#pragma unroll
    for (int nt = 0; nt < 4; ++nt)
        acc[nt] = __builtin_amdgcn_wmma_f32_16x16x32_f16(false, a1, false, b1[nt],
                                                         (short)0, acc[nt], false, false);

    // C tile layout: VGPR v -> row v + 8*lhalf, col = lane%16
#pragma unroll
    for (int nt = 0; nt < 4; ++nt) {
        int n = (ntb + nt) * 16 + lmod;
#pragma unroll
        for (int v = 0; v < 8; ++v) {
            int row = mt * 16 + v + 8 * lhalf;
            C[row * N + n] = acc[nt][v];
        }
    }
}

// ---------------------------------------------------------------------------
// LSTM recurrence, one block (1024 thr = 32 waves) per direction (gridDim.x=2).
// Per step: 32 waves x 4 row-tiles x 16 k-steps of WMMA:
//   z[2048] = gates_pre[t] + W_hh @ h   (accumulator seeded with gates_pre)
// h lives in LDS as f16 (broadcast into the WMMA B operand); c in registers.
// W_hh pre-packed f16 A-layout -> contiguous 32B/lane global_load_b128 pairs,
// prefetched one k-step ahead (fully unrolled -> SSA renaming, no movs).
// Output h written directly in A-packed layout for the next GEMM (KT=32).
// ---------------------------------------------------------------------------
__global__ __launch_bounds__(1024)
void lstm_rec_kernel(const float* __restrict__ gates_f, const float* __restrict__ gates_b,
                     const _Float16* __restrict__ whh_f, const _Float16* __restrict__ whh_b,
                     _Float16* __restrict__ h_pk /* A-packed [512 rows][1024 cols] */) {
    __shared__ __align__(32) _Float16 hlds[H_DIM];
    __shared__ float glds[G_DIM];

    int dir = blockIdx.x;
    const float*    gates  = dir ? gates_b : gates_f;
    const _Float16* pk     = dir ? whh_b   : whh_f;
    int             coloff = dir ? H_DIM   : 0;

    int tid   = threadIdx.x;
    int l     = tid & 31;
    int w     = tid >> 5;
    int lhalf = l >> 4;

    float c = 0.0f;
    if (tid < H_DIM) hlds[tid] = (_Float16)0.0f;
    __syncthreads();

    for (int t = 0; t < L_SEQ; ++t) {
        int tt = dir ? (L_SEQ - 1 - t) : t;
        const float* gp = gates + (size_t)tt * G_DIM;

#pragma unroll
        for (int tile = 0; tile < 4; ++tile) {
            int rt    = w * 4 + tile;      // 0..127 row tiles of W_hh
            int rbase = rt * 16;
            const _Float16* tp = pk + (((size_t)rt << 4) << 9) + (l << 4);
            v8f acc;
#pragma unroll
            for (int v = 0; v < 8; ++v) acc[v] = gp[rbase + 8 * lhalf + v];

            v16h a_cur = *(const v16h*)(tp);
#pragma unroll
            for (int kt = 0; kt < 15; ++kt) {
                v16h a_nxt = *(const v16h*)(tp + ((kt + 1) << 9));
                v16h b = *(const v16h*)(&hlds[kt * 32 + (lhalf << 4)]);
                acc = __builtin_amdgcn_wmma_f32_16x16x32_f16(false, a_cur, false, b,
                                                             (short)0, acc, false, false);
                a_cur = a_nxt;
            }
            {
                v16h b = *(const v16h*)(&hlds[15 * 32 + (lhalf << 4)]);
                acc = __builtin_amdgcn_wmma_f32_16x16x32_f16(false, a_cur, false, b,
                                                             (short)0, acc, false, false);
            }
            // all 16 columns identical; lanes 0 and 16 hold rows 0-7 / 8-15
            if ((l & 15) == 0) {
#pragma unroll
                for (int v = 0; v < 8; ++v) glds[rbase + 8 * lhalf + v] = acc[v];
            }
        }
        __syncthreads();

        if (tid < H_DIM) {
            float zi = glds[tid];
            float zf = glds[H_DIM + tid];
            float zg = glds[2 * H_DIM + tid];
            float zo = glds[3 * H_DIM + tid];
            float i = fast_sigmoid(zi);
            float f = fast_sigmoid(zf);
            float g = fast_tanh(zg);
            float o = fast_sigmoid(zo);
            c = f * c + i * g;
            float h = o * fast_tanh(c);
            hlds[tid] = (_Float16)h;
            h_pk[packA_idx(tt, coloff + tid, DIN1 / 32)] = (_Float16)h;
        }
        __syncthreads();
    }
}

// ---------------------------------------------------------------------------
// Pairwise scores: scores[i][j] = out_b + sum_m out_w[m]*tanh(mlp[i][m]+mlp[j+1][m])
// 16x16 (i,j) tile per 256-thread block; mlp rows + out_w staged in LDS in
// 128-column chunks. Dominant cost: 134M tanh (v_exp_f32 TRANS pipe).
// ---------------------------------------------------------------------------
__global__ __launch_bounds__(256)
void pair_kernel(const float* __restrict__ mlp, const float* __restrict__ out_w,
                 const float* __restrict__ out_b, float* __restrict__ scores) {
    __shared__ float Ai[16][128];
    __shared__ float Bj[16][128];
    __shared__ float wc[128];
    int tid = threadIdx.x;
    int ti  = tid & 15;
    int tj  = tid >> 4;
    int i   = blockIdx.x * 16 + ti;
    int j   = blockIdx.y * 16 + tj;      // child index, valid if < 511
    float acc = out_b[0];
    for (int ch = 0; ch < 4; ++ch) {
        __syncthreads();
#pragma unroll
        for (int s = 0; s < 8; ++s) {
            int flat = s * 256 + tid;
            int r = flat >> 7, cc = flat & 127;
            Ai[r][cc] = mlp[(blockIdx.x * 16 + r) * M_DIM + ch * 128 + cc];
            int rj = blockIdx.y * 16 + r + 1;
            Bj[r][cc] = (rj < L_SEQ) ? mlp[rj * M_DIM + ch * 128 + cc] : 0.0f;
        }
        if (tid < 128) wc[tid] = out_w[ch * 128 + tid];
        __syncthreads();
#pragma unroll 4
        for (int cc = 0; cc < 128; ++cc) {
            float s = Ai[ti][cc] + Bj[tj][cc];
            acc = fmaf(wc[cc], fast_tanh(s), acc);
        }
    }
    if (j < (L_SEQ - 1)) scores[i * (L_SEQ - 1) + j] = acc;
}

// ---------------------------------------------------------------------------
extern "C" void kernel_launch(void* const* d_in, const int* in_sizes, int n_in,
                              void* d_out, int out_size, void* d_ws, size_t ws_size,
                              hipStream_t stream) {
    (void)in_sizes; (void)n_in; (void)out_size; (void)ws_size;

    const int*   word_idx = (const int*)d_in[0];
    const int*   pos_idx  = (const int*)d_in[1];
    const float* w_embed  = (const float*)d_in[2];
    const float* p_embed  = (const float*)d_in[3];
    const float* W_ih_l0  = (const float*)d_in[4];
    const float* W_hh_l0  = (const float*)d_in[5];
    const float* b_ih_l0  = (const float*)d_in[6];
    const float* b_hh_l0  = (const float*)d_in[7];
    const float* W_ih_l0r = (const float*)d_in[8];
    const float* W_hh_l0r = (const float*)d_in[9];
    const float* b_ih_l0r = (const float*)d_in[10];
    const float* b_hh_l0r = (const float*)d_in[11];
    const float* W_ih_l1  = (const float*)d_in[12];
    const float* W_hh_l1  = (const float*)d_in[13];
    const float* b_ih_l1  = (const float*)d_in[14];
    const float* b_hh_l1  = (const float*)d_in[15];
    const float* W_ih_l1r = (const float*)d_in[16];
    const float* W_hh_l1r = (const float*)d_in[17];
    const float* b_ih_l1r = (const float*)d_in[18];
    const float* b_hh_l1r = (const float*)d_in[19];
    const float* mlp_W    = (const float*)d_in[20];
    const float* mlp_b    = (const float*)d_in[21];
    const float* out_w    = (const float*)d_in[22];
    const float* out_b    = (const float*)d_in[23];

    // ---- workspace carve-up (all 256B aligned) ----
    char* ws = (char*)d_ws;
    size_t off = 0;
    auto alloc = [&](size_t bytes) -> char* {
        char* p = ws + off;
        off += (bytes + 255) & ~(size_t)255;
        return p;
    };
    const size_t WHH_HALVES = (size_t)G_DIM * H_DIM;          // per matrix
    _Float16* whh_pk  = (_Float16*)alloc(4 * WHH_HALVES * sizeof(_Float16)); // 8 MB
    _Float16* xcat_pk = (_Float16*)alloc((size_t)L_SEQ * DIN0 * sizeof(_Float16));
    _Float16* h0_pk   = (_Float16*)alloc((size_t)L_SEQ * DIN1 * sizeof(_Float16));
    _Float16* h1_pk   = (_Float16*)alloc((size_t)L_SEQ * DIN1 * sizeof(_Float16));
    float*    gates_f = (float*)alloc((size_t)L_SEQ * G_DIM * sizeof(float)); // 4 MB
    float*    gates_b = (float*)alloc((size_t)L_SEQ * G_DIM * sizeof(float)); // 4 MB
    float*    mlp     = (float*)alloc((size_t)L_SEQ * M_DIM * sizeof(float)); // 1 MB
    _Float16* wih0h   = (_Float16*)alloc((size_t)G_DIM * DIN0 * sizeof(_Float16));
    _Float16* wih0rh  = (_Float16*)alloc((size_t)G_DIM * DIN0 * sizeof(_Float16));
    _Float16* wih1h   = (_Float16*)alloc((size_t)G_DIM * DIN1 * sizeof(_Float16));
    _Float16* wih1rh  = (_Float16*)alloc((size_t)G_DIM * DIN1 * sizeof(_Float16));
    _Float16* mlpWh   = (_Float16*)alloc((size_t)M_DIM * DIN1 * sizeof(_Float16));

    // 0) one-time f16 weight copies (B operand: row-major f16 -> contiguous b128)
    auto cvt = [&](const float* s, _Float16* d, int n) {
        f32_to_f16_kernel<<<(n + 255) / 256, 256, 0, stream>>>(s, d, n);
    };
    cvt(W_ih_l0,  wih0h,  G_DIM * DIN0);
    cvt(W_ih_l0r, wih0rh, G_DIM * DIN0);
    cvt(W_ih_l1,  wih1h,  G_DIM * DIN1);
    cvt(W_ih_l1r, wih1rh, G_DIM * DIN1);
    cvt(mlp_W,    mlpWh,  M_DIM * DIN1);

    // 1) pack all four W_hh into WMMA A-operand f16 layout (L2-resident, reused 512x)
    pack_whh_kernel<<<dim3((G_DIM * H_DIM) / 256, 4), 256, 0, stream>>>(
        W_hh_l0, W_hh_l0r, W_hh_l1, W_hh_l1r, whh_pk);

    // 2) embedding gather -> A-packed f16 activations
    embed_kernel<<<L_SEQ, DIN0, 0, stream>>>(word_idx, pos_idx, w_embed, p_embed, xcat_pk);

    // 3) layer-0 input-gate GEMMs: [512,320] x [320,2048] + (b_ih+b_hh)
    gemm_wmma_kernel<<<dim3(4, G_DIM / 64), 256, 0, stream>>>(
        xcat_pk, wih0h, b_ih_l0, b_hh_l0, gates_f, DIN0, G_DIM);
    gemm_wmma_kernel<<<dim3(4, G_DIM / 64), 256, 0, stream>>>(
        xcat_pk, wih0rh, b_ih_l0r, b_hh_l0r, gates_b, DIN0, G_DIM);

    // 4) layer-0 recurrence (fwd + bwd concurrently)
    lstm_rec_kernel<<<2, 1024, 0, stream>>>(gates_f, gates_b,
                                            whh_pk + 0 * WHH_HALVES,
                                            whh_pk + 1 * WHH_HALVES, h0_pk);

    // 5) layer-1 input-gate GEMMs: [512,1024] x [1024,2048]
    gemm_wmma_kernel<<<dim3(4, G_DIM / 64), 256, 0, stream>>>(
        h0_pk, wih1h, b_ih_l1, b_hh_l1, gates_f, DIN1, G_DIM);
    gemm_wmma_kernel<<<dim3(4, G_DIM / 64), 256, 0, stream>>>(
        h0_pk, wih1rh, b_ih_l1r, b_hh_l1r, gates_b, DIN1, G_DIM);

    // 6) layer-1 recurrence
    lstm_rec_kernel<<<2, 1024, 0, stream>>>(gates_f, gates_b,
                                            whh_pk + 2 * WHH_HALVES,
                                            whh_pk + 3 * WHH_HALVES, h1_pk);

    // 7) MLP projection: [512,1024] x [1024,512] + mlp_b
    gemm_wmma_kernel<<<dim3(4, M_DIM / 64), 256, 0, stream>>>(
        h1_pk, mlpWh, mlp_b, nullptr, mlp, DIN1, M_DIM);

    // 8) pairwise tanh scores -> d_out [512,511]
    pair_kernel<<<dim3(L_SEQ / 16, L_SEQ / 16), 256, 0, stream>>>(
        mlp, out_w, out_b, (float*)d_out);
}
